// Quantize_26740466384906
// MI455X (gfx1250) — matrix operand order repaired and checked by reading, not compiled, not run on hardware
//
#include <hip/hip_runtime.h>

typedef _Float16 v2h  __attribute__((ext_vector_type(2)));
typedef _Float16 v8h  __attribute__((ext_vector_type(8)));
typedef _Float16 v16h __attribute__((ext_vector_type(16)));
typedef float    v8f  __attribute__((ext_vector_type(8)));

#define N_TOK    32768          // B*H = 8*4096 tokens
#define DIM      512            // C / GROUPS
#define N_EMB    1024           // codebook size
#define ZQ_ELEMS (N_TOK * DIM)  // 16,777,216

// ---------------------------------------------------------------------------
// K0: embed_w f32 -> f16 copy (for WMMA) + per-code squared norm (f32).
// ---------------------------------------------------------------------------
__global__ __launch_bounds__(256) void k_embed_prep(const float* __restrict__ ew,
                                                    _Float16* __restrict__ ewH,
                                                    float* __restrict__ eNorm) {
    const int code = blockIdx.x;
    const int tid  = threadIdx.x;
    const float2 v = ((const float2*)(ew + (size_t)code * DIM))[tid];
    v2h h; h.x = (_Float16)v.x; h.y = (_Float16)v.y;
    ((v2h*)(ewH + (size_t)code * DIM))[tid] = h;
    float s = v.x * v.x + v.y * v.y;
    #pragma unroll
    for (int off = 16; off; off >>= 1) s += __shfl_down(s, off, 32);
    __shared__ float ls[8];
    if ((tid & 31) == 0) ls[tid >> 5] = s;
    __syncthreads();
    if (tid == 0) {
        float t = 0.f;
        #pragma unroll
        for (int i = 0; i < 8; ++i) t += ls[i];
        eNorm[code] = t;
    }
}

// Convert 16 f32 (A layout pattern) to one v16h A fragment.
// p points at this lane's K-base; elems 0..7 <- p[0..7], 8..15 <- p[16..23].
__device__ __forceinline__ v16h load_a_chunk(const float* __restrict__ p) {
    float4 f0 = *(const float4*)(p);
    float4 f1 = *(const float4*)(p + 4);
    float4 f2 = *(const float4*)(p + 16);
    float4 f3 = *(const float4*)(p + 20);
    v16h t;
    t[0]  = (_Float16)f0.x; t[1]  = (_Float16)f0.y;
    t[2]  = (_Float16)f0.z; t[3]  = (_Float16)f0.w;
    t[4]  = (_Float16)f1.x; t[5]  = (_Float16)f1.y;
    t[6]  = (_Float16)f1.z; t[7]  = (_Float16)f1.w;
    t[8]  = (_Float16)f2.x; t[9]  = (_Float16)f2.y;
    t[10] = (_Float16)f2.z; t[11] = (_Float16)f2.w;
    t[12] = (_Float16)f3.x; t[13] = (_Float16)f3.y;
    t[14] = (_Float16)f3.z; t[15] = (_Float16)f3.w;
    return t;
}

// ---------------------------------------------------------------------------
// K2: fused WMMA distance + argmin.
// 8 waves/WG; each wave owns TWO 16-token tiles (32 tokens, 2x128 VGPR A
// panel) and sweeps all 1024 codes. Each B fragment (1 KB) feeds two
// independent v_wmma_f32_16x16x32_f16 chains -> halves B-operand traffic
// (the bottleneck) and doubles per-wave WMMA ILP.
// dist = ||e||^2 - 2*(z.e)  (||z||^2 constant per token -> irrelevant).
// ---------------------------------------------------------------------------
__global__ __launch_bounds__(256) void k_argmin(const float* __restrict__ z,
                                                const _Float16* __restrict__ ewH,
                                                const float* __restrict__ eNorm,
                                                int* __restrict__ idxOut,
                                                float* __restrict__ indF) {
    const int tid     = threadIdx.x;
    const int lane    = tid & 31;
    const int wave    = tid >> 5;
    const int tokBase = blockIdx.x * 256 + wave * 32;   // 32 tokens per wave
    const int hi      = (lane < 16) ? 0 : 1;            // which half-wave
    const int l15     = lane & 15;

    // ---- resident A panels: ISA 16-bit A 16x32 layout ----
    // lanes 0-15 : M=l,    elems 0..7 -> K=c*32+{0..7}, 8..15 -> K=c*32+{16..23}
    // lanes16-31 : M=l-16, same pattern shifted by +8 in K
    v16h a0[16], a1[16];
    {
        const float* zrow0 = z + (size_t)(tokBase + l15) * DIM + (hi ? 8 : 0);
        const float* zrow1 = zrow0 + (size_t)16 * DIM;
        #pragma unroll
        for (int c = 0; c < 16; ++c) {
            a0[c] = load_a_chunk(zrow0 + c * 32);
            a1[c] = load_a_chunk(zrow1 + c * 32);
        }
    }

    float bestVal0[8], bestVal1[8];
    int   bestIdx0[8], bestIdx1[8];
    #pragma unroll
    for (int r = 0; r < 8; ++r) {
        bestVal0[r] = 3.4e38f; bestIdx0[r] = 0;
        bestVal1[r] = 3.4e38f; bestIdx1[r] = 0;
    }

    // ---- B base: ISA 16-bit B 32x16 layout ----
    // lanes 0-15: N=l, elems 0..7 -> K=c*32+{0..7}, 8..15 -> K=c*32+{16..23}; lanes 16-31: +8 in K
    const _Float16* brow = ewH + (size_t)l15 * DIM + (hi ? 8 : 0);

    for (int nb = 0; nb < N_EMB; nb += 16) {
        v8f acc0 = {};
        v8f acc1 = {};
        const _Float16* bp = brow + (size_t)nb * DIM;
        #pragma unroll
        for (int c = 0; c < 16; ++c) {
            v8h lo = *(const v8h*)(bp + c * 32);
            v8h hh = *(const v8h*)(bp + c * 32 + 16);
            v16h b = __builtin_shufflevector(lo, hh, 0, 1, 2, 3, 4, 5, 6, 7,
                                                     8, 9, 10, 11, 12, 13, 14, 15);
            acc0 = __builtin_amdgcn_wmma_f32_16x16x32_f16(
                false, a0[c], false, b, (short)0, acc0, false, false);
            acc1 = __builtin_amdgcn_wmma_f32_16x16x32_f16(
                false, a1[c], false, b, (short)0, acc1, false, false);
        }
        // C/D layout: acc[r] -> token M = r + 8*hi, code N = l15
        const int   code = nb + l15;
        const float en   = eNorm[code];
        #pragma unroll
        for (int r = 0; r < 8; ++r) {
            float d0 = en - 2.0f * acc0[r];
            if (d0 < bestVal0[r]) { bestVal0[r] = d0; bestIdx0[r] = code; }
            float d1 = en - 2.0f * acc1[r];
            if (d1 < bestVal1[r]) { bestVal1[r] = d1; bestIdx1[r] = code; }
        }
    }

    // cross-lane argmin within each 16-lane half (xor stays inside the half)
    #pragma unroll
    for (int r = 0; r < 8; ++r) {
        float v0 = bestVal0[r]; int i0 = bestIdx0[r];
        float v1 = bestVal1[r]; int i1 = bestIdx1[r];
        #pragma unroll
        for (int off = 8; off; off >>= 1) {
            float ov0 = __shfl_xor(v0, off, 32);
            int   oi0 = __shfl_xor(i0, off, 32);
            if (ov0 < v0 || (ov0 == v0 && oi0 < i0)) { v0 = ov0; i0 = oi0; }
            float ov1 = __shfl_xor(v1, off, 32);
            int   oi1 = __shfl_xor(i1, off, 32);
            if (ov1 < v1 || (ov1 == v1 && oi1 < i1)) { v1 = ov1; i1 = oi1; }
        }
        if (l15 == 0) {
            const int t0 = tokBase + r + hi * 8;
            idxOut[t0] = i0;
            indF[t0]   = (float)i0;
            const int t1 = t0 + 16;
            idxOut[t1] = i1;
            indF[t1]   = (float)i1;
        }
    }
}

// ---------------------------------------------------------------------------
// K3: gather z_q = embed_w[ind] (f32, exact) + per-token partial sum of
// (z_q - z)^2. One block per token.
// ---------------------------------------------------------------------------
__global__ __launch_bounds__(256) void k_gather(const float* __restrict__ z,
                                                const float* __restrict__ ew,
                                                const int* __restrict__ idx,
                                                float* __restrict__ zq,
                                                float* __restrict__ partials) {
    const int token = blockIdx.x;
    const int tid   = threadIdx.x;
    const int code  = idx[token];
    const float2 e  = ((const float2*)(ew + (size_t)code  * DIM))[tid];
    const float2 zz = ((const float2*)(z  + (size_t)token * DIM))[tid];
    ((float2*)(zq + (size_t)token * DIM))[tid] = e;
    const float dx = e.x - zz.x, dy = e.y - zz.y;
    float s = dx * dx + dy * dy;
    #pragma unroll
    for (int off = 16; off; off >>= 1) s += __shfl_down(s, off, 32);
    __shared__ float ls[8];
    if ((tid & 31) == 0) ls[tid >> 5] = s;
    __syncthreads();
    if (tid == 0) {
        float t = 0.f;
        #pragma unroll
        for (int i = 0; i < 8; ++i) t += ls[i];
        partials[token] = t;
    }
}

// ---------------------------------------------------------------------------
// K4: deterministic final reduction -> diff = 10*(0.25+1)*MSE.
// ---------------------------------------------------------------------------
__global__ __launch_bounds__(256) void k_diff(const float* __restrict__ partials,
                                              float* __restrict__ diffOut) {
    const int tid = threadIdx.x;
    float s = 0.f;
    for (int i = tid; i < N_TOK; i += 256) s += partials[i];
    #pragma unroll
    for (int off = 16; off; off >>= 1) s += __shfl_down(s, off, 32);
    __shared__ float ls[8];
    if ((tid & 31) == 0) ls[tid >> 5] = s;
    __syncthreads();
    if (tid == 0) {
        float t = 0.f;
        #pragma unroll
        for (int i = 0; i < 8; ++i) t += ls[i];
        *diffOut = 10.0f * 1.25f * t / (float)ZQ_ELEMS;
    }
}

// ---------------------------------------------------------------------------
extern "C" void kernel_launch(void* const* d_in, const int* in_sizes, int n_in,
                              void* d_out, int out_size, void* d_ws, size_t ws_size,
                              hipStream_t stream) {
    const float* z  = (const float*)d_in[0];   // [8,4096,512] f32
    const float* ew = (const float*)d_in[1];   // [1024,512]   f32

    float* out  = (float*)d_out;
    float* zq   = out;                           // 16,777,216 floats
    float* diff = out + (size_t)ZQ_ELEMS;        // 1 float
    float* indF = out + (size_t)ZQ_ELEMS + 1;    // 32768 floats (indices)

    char* ws = (char*)d_ws;
    _Float16* ewH      = (_Float16*)ws;                                   // 1 MB
    float*    eNorm    = (float*)(ws + (size_t)N_EMB * DIM * 2);          // 4 KB
    int*      idx      = (int*)((char*)eNorm + (size_t)N_EMB * 4);        // 128 KB
    float*    partials = (float*)((char*)idx + (size_t)N_TOK * 4);        // 128 KB

    k_embed_prep<<<N_EMB,       256, 0, stream>>>(ew, ewH, eNorm);
    k_argmin    <<<N_TOK / 256, 256, 0, stream>>>(z, ewH, eNorm, idx, indF);
    k_gather    <<<N_TOK,       256, 0, stream>>>(z, ew, idx, zq, partials);
    k_diff      <<<1,           256, 0, stream>>>(partials, diff);
}
